// BahdanauAttention_3427383902426
// MI455X (gfx1250) — matrix operand order repaired and checked by reading, not compiled, not run on hardware
//
#include <hip/hip_runtime.h>

// Problem dimensions
#define L_DIM 4096
#define N_DIM 32
#define E_DIM 1024
#define D_DIM 1024
#define A_DIM 512
#define M_DIM (L_DIM * N_DIM)   // 131072 rows of enc viewed as (L*N, E)

// CDNA5 WMMA vector types
typedef __attribute__((ext_vector_type(16))) __bf16        v16bf;
typedef __attribute__((ext_vector_type(2)))  __bf16        bf16x2;
typedef __attribute__((ext_vector_type(8)))  float         v8f;
typedef __attribute__((ext_vector_type(4)))  float         f32x4;
typedef __attribute__((ext_vector_type(2)))  float         f32x2;
typedef __attribute__((ext_vector_type(4)))  unsigned int  u32x4;

// fp32 pair -> packed bf16 dword, RNE, via hardware v_cvt_pk_bf16_f32
__device__ __forceinline__ unsigned int pack2_bf16(float a, float b) {
  f32x2 v;
  v.x = a;
  v.y = b;
  bf16x2 h = __builtin_convertvector(v, bf16x2);
  return __builtin_bit_cast(unsigned int, h);
}

// hardware V_TANH_F32 (TRANS op) when the builtin exists, else ocml
__device__ __forceinline__ float fast_tanh(float x) {
#if __has_builtin(__builtin_amdgcn_tanhf)
  return __builtin_amdgcn_tanhf(x);
#else
  return tanhf(x);
#endif
}

// ---------------------------------------------------------------------------
// Kernel 1: proj_dec = dec (32x1024) @ W_a^T (1024x512)  -> (32,512) fp32
// ---------------------------------------------------------------------------
__global__ void __launch_bounds__(256)
proj_dec_kernel(const float* __restrict__ dec, const float* __restrict__ Wa,
                float* __restrict__ out) {
  const int idx = blockIdx.x * 256 + threadIdx.x;   // 16384 threads
  const int n = idx >> 9;
  const int a = idx & 511;
  const float* dr = dec + (size_t)n * D_DIM;
  const float* wr = Wa  + (size_t)a * D_DIM;
  float acc = 0.f;
  for (int k = 0; k < D_DIM; k += 4) {
    f32x4 d = *(const f32x4*)(dr + k);
    f32x4 w = *(const f32x4*)(wr + k);
    acc += d.x * w.x + d.y * w.y + d.z * w.z + d.w * w.w;
  }
  out[idx] = acc;   // layout: n*512 + a
}

// ---------------------------------------------------------------------------
// Kernel 2: U_a (512x1024 fp32) -> bf16, row-major (K contiguous)
// ---------------------------------------------------------------------------
__global__ void __launch_bounds__(256)
cvt_bf16_kernel(const float* __restrict__ src, unsigned short* __restrict__ dst) {
  const int idx = blockIdx.x * 256 + threadIdx.x;   // 65536 threads x 8 elems
  const size_t base = (size_t)idx * 8;
  f32x4 a = *(const f32x4*)(src + base);
  f32x4 b = *(const f32x4*)(src + base + 4);
  u32x4 q;
  q.x = pack2_bf16(a.x, a.y);
  q.y = pack2_bf16(a.z, a.w);
  q.z = pack2_bf16(b.x, b.y);
  q.w = pack2_bf16(b.z, b.w);
  *(u32x4*)(dst + base) = q;
}

// ---------------------------------------------------------------------------
// Kernel 3: main fused GEMM + tanh + v_a-dot  (bf16 WMMA, fp32 accum)
//   scores_part[colTile][m] = sum_{a in 64-col slice} tanh(C[m,a]+pd[n,a])*v_a[a]
// Block: 256 threads (8 waves). Tile BM=256 x BN=64, BK=32 (32 K-steps).
// Wave w owns rows [w*32, w*32+32) x all 64 cols: 2x4 fragments of 16x16.
// ---------------------------------------------------------------------------
#define LDS_PITCH 80              // 64B bf16 row + 16B pad -> 20-dword stride
#define LDS_BUF   (256 * LDS_PITCH)

__global__ void __launch_bounds__(256)
attn_score_gemm(const float* __restrict__ enc,
                const float* __restrict__ projdec,
                const unsigned short* __restrict__ Ubf,
                const float* __restrict__ va,
                float* __restrict__ spart) {
  __shared__ __align__(16) unsigned char smem[2 * LDS_BUF];

  const int t       = threadIdx.x;
  const int colTile = blockIdx.x;            // 0..7   (fastest: L2 reuse of enc)
  const int m0      = blockIdx.y * 256;      // 0..511 tiles
  const int colbase = colTile * 64;

  const int wid  = t >> 5;
  const int lane = t & 31;
  const int lr   = lane & 15;                // row/col within 16
  const int lh   = lane >> 4;                // half select

  const v8f vzero = {0.f, 0.f, 0.f, 0.f, 0.f, 0.f, 0.f, 0.f};
  v8f acc[2][4];
#pragma unroll
  for (int mi = 0; mi < 2; ++mi)
#pragma unroll
    for (int ni = 0; ni < 4; ++ni) acc[mi][ni] = vzero;

  // each thread stages one 256-row-tile row: 32 fp32 (one full 128B line)
  const float* arow = enc + (size_t)(m0 + t) * E_DIM;
  f32x4 stage[8];
#pragma unroll
  for (int j = 0; j < 8; ++j) stage[j] = *(const f32x4*)(arow + j * 4);

  int buf = 0;
  for (int kt = 0; kt < E_DIM / 32; ++kt) {
    // ---- convert fp32 -> bf16 (v_cvt_pk_bf16_f32), store A tile row to LDS
    unsigned char* sb = smem + buf * LDS_BUF + t * LDS_PITCH;
#pragma unroll
    for (int j = 0; j < 4; ++j) {
      u32x4 q;
      q.x = pack2_bf16(stage[2 * j].x,     stage[2 * j].y);
      q.y = pack2_bf16(stage[2 * j].z,     stage[2 * j].w);
      q.z = pack2_bf16(stage[2 * j + 1].x, stage[2 * j + 1].y);
      q.w = pack2_bf16(stage[2 * j + 1].z, stage[2 * j + 1].w);
      *(u32x4*)(sb + j * 16) = q;
    }
    __syncthreads();

    // ---- prefetch next K-slab while this one computes (latency hiding)
    if (kt + 1 < E_DIM / 32) {
      const float* p = arow + (kt + 1) * 32;
#pragma unroll
      for (int j = 0; j < 8; ++j) stage[j] = *(const f32x4*)(p + j * 4);
    }

    // ---- A fragments from LDS (ISA 16-bit A layout: lane m -> K{0-7,16-23},
    //      lane m+16 -> K{8-15,24-31}; i.e. byte chunks lh*16 and lh*16+32)
    v16bf afrag[2], bfrag[4];
    const unsigned char* ab = smem + buf * LDS_BUF + lh * 16;
#pragma unroll
    for (int mi = 0; mi < 2; ++mi) {
      const unsigned char* ap = ab + (wid * 32 + mi * 16 + lr) * LDS_PITCH;
      union { u32x4 q[2]; v16bf v; } u;
      u.q[0] = *(const u32x4*)(ap);
      u.q[1] = *(const u32x4*)(ap + 32);
      afrag[mi] = u.v;
    }
    // ---- B fragments straight from bf16 U_a (lane = column a; lanes 0-15 ->
    //      K{0-7,16-23}, lanes 16-31 -> K{8-15,24-31})
    const int kg = kt * 32 + lh * 8;
#pragma unroll
    for (int ni = 0; ni < 4; ++ni) {
      const int a = colbase + ni * 16 + lr;
      const u32x4* bq = (const u32x4*)(Ubf + (size_t)a * E_DIM + kg);
      union { u32x4 q[2]; v16bf v; } u;
      u.q[0] = bq[0];
      u.q[1] = bq[2];
      bfrag[ni] = u.v;
    }

#pragma unroll
    for (int mi = 0; mi < 2; ++mi)
#pragma unroll
      for (int ni = 0; ni < 4; ++ni)
        acc[mi][ni] = __builtin_amdgcn_wmma_f32_16x16x32_bf16(
            false, afrag[mi], false, bfrag[ni], (short)0, acc[mi][ni],
            false, false);
    buf ^= 1;
  }

  // ---- epilogue: tanh(C + proj_dec[n]) . v_a  reduced over this 64-col slice
  float vva[4];
#pragma unroll
  for (int ni = 0; ni < 4; ++ni) vva[ni] = va[colbase + ni * 16 + lr];

#pragma unroll
  for (int mi = 0; mi < 2; ++mi) {
#pragma unroll
    for (int r = 0; r < 8; ++r) {
      // C layout: element r = row (mi*16 + lh*8 + r), col = colbase+ni*16+lr
      const int nb = mi * 16 + lh * 8 + r;   // batch n = row % 32 (m0,wid ≡ 0 mod 32)
      float s = 0.f;
#pragma unroll
      for (int ni = 0; ni < 4; ++ni) {
        const int a = colbase + ni * 16 + lr;
        const float e = fast_tanh(acc[mi][ni][r] + projdec[nb * A_DIM + a]);
        s += e * vva[ni];
      }
      // reduce the 16 lanes of each half-wave (rows differ between halves)
      s += __shfl_xor(s, 1, 32);
      s += __shfl_xor(s, 2, 32);
      s += __shfl_xor(s, 4, 32);
      s += __shfl_xor(s, 8, 32);
      if (lr == 0) {
        const int mrow = m0 + wid * 32 + mi * 16 + lh * 8 + r;
        spart[(size_t)colTile * M_DIM + mrow] = s;
      }
    }
  }
}

// ---------------------------------------------------------------------------
// Kernel 4: combine 8 score partials, softmax over L per batch n
// ---------------------------------------------------------------------------
__global__ void __launch_bounds__(256)
softmax_kernel(const float* __restrict__ spart, float* __restrict__ wout) {
  const int n = blockIdx.x;       // 32 blocks
  const int t = threadIdx.x;      // 256 threads, 16 l's each
  __shared__ float red[256];
  float sv[16];
  float mx = -3.402823466e38f;
#pragma unroll
  for (int i = 0; i < 16; ++i) {
    const int l = i * 256 + t;
    const size_t m = (size_t)l * N_DIM + n;
    float s = 0.f;
#pragma unroll
    for (int p = 0; p < 8; ++p) s += spart[(size_t)p * M_DIM + m];
    sv[i] = s;
    mx = fmaxf(mx, s);
  }
  red[t] = mx; __syncthreads();
  for (int s = 128; s > 0; s >>= 1) {
    if (t < s) red[t] = fmaxf(red[t], red[t + s]);
    __syncthreads();
  }
  const float bm = red[0];
  __syncthreads();
  float lsum = 0.f;
#pragma unroll
  for (int i = 0; i < 16; ++i) { sv[i] = __expf(sv[i] - bm); lsum += sv[i]; }
  red[t] = lsum; __syncthreads();
  for (int s = 128; s > 0; s >>= 1) {
    if (t < s) red[t] += red[t + s];
    __syncthreads();
  }
  const float inv = 1.0f / red[0];
#pragma unroll
  for (int i = 0; i < 16; ++i)
    wout[(size_t)n * L_DIM + i * 256 + t] = sv[i] * inv;
}

// ---------------------------------------------------------------------------
// Kernel 5: context partials over L chunks (second streaming pass of enc)
// grid (4 e-chunks, 32 n, 8 l-chunks) x 256 threads -> coalesced 1KB/l rows
// ---------------------------------------------------------------------------
__global__ void __launch_bounds__(256)
ctx_partial_kernel(const float* __restrict__ w, const float* __restrict__ enc,
                   float* __restrict__ part) {
  const int e  = blockIdx.x * 256 + threadIdx.x;
  const int n  = blockIdx.y;
  const int lc = blockIdx.z;
  const int l0 = lc * 512;
  float acc = 0.f;
  for (int l = l0; l < l0 + 512; ++l)
    acc += w[(size_t)n * L_DIM + l] * enc[((size_t)l * N_DIM + n) * E_DIM + e];
  part[(size_t)lc * (N_DIM * E_DIM) + (size_t)n * E_DIM + e] = acc;
}

// ---------------------------------------------------------------------------
// Kernel 6: reduce 8 context partials -> d_out[0 .. 32767]
// ---------------------------------------------------------------------------
__global__ void __launch_bounds__(256)
ctx_reduce_kernel(const float* __restrict__ part, float* __restrict__ ctx) {
  const int i = blockIdx.x * 256 + threadIdx.x;   // 32768
  float acc = 0.f;
#pragma unroll
  for (int p = 0; p < 8; ++p) acc += part[(size_t)p * (N_DIM * E_DIM) + i];
  ctx[i] = acc;
}

// ---------------------------------------------------------------------------
extern "C" void kernel_launch(void* const* d_in, const int* in_sizes, int n_in,
                              void* d_out, int out_size, void* d_ws, size_t ws_size,
                              hipStream_t stream) {
  (void)in_sizes; (void)n_in; (void)out_size; (void)ws_size;
  const float* dec = (const float*)d_in[0];   // (N, D)
  const float* enc = (const float*)d_in[1];   // (L, N, E)
  const float* Wa  = (const float*)d_in[2];   // (A, D)
  const float* Ua  = (const float*)d_in[3];   // (A, E)
  const float* va  = (const float*)d_in[4];   // (A,)
  float* out = (float*)d_out;                 // [context (N*E) | weights (N*L)]
  char*  ws  = (char*)d_ws;

  // workspace layout (bytes)
  float*          projdec = (float*)(ws);                              //  64 KB
  unsigned short* Ubf     = (unsigned short*)(ws + 65536);             //   1 MB
  float*          spart   = (float*)(ws + 65536 + 1048576);            //   4 MB
  float*          cpart   = (float*)(ws + 65536 + 1048576 + 4194304);  //   1 MB

  float* wout = out + (size_t)N_DIM * E_DIM;  // attention weights region

  proj_dec_kernel  <<<64,              256, 0, stream>>>(dec, Wa, projdec);
  cvt_bf16_kernel  <<<256,             256, 0, stream>>>(Ua, Ubf);
  attn_score_gemm  <<<dim3(8, 512),    256, 0, stream>>>(enc, projdec, Ubf, va, spart);
  softmax_kernel   <<<32,              256, 0, stream>>>(spart, wout);
  ctx_partial_kernel<<<dim3(4, 32, 8), 256, 0, stream>>>(wout, enc, cpart);
  ctx_reduce_kernel<<<128,             256, 0, stream>>>(cpart, out);
}
